// DecoderLayer_22050362098049
// MI455X (gfx1250) — matrix operand order repaired and checked by reading, not compiled, not run on hardware
//
#include <hip/hip_runtime.h>
#include <math.h>

// ---------------------------------------------------------------------------
// Types for CDNA5 WMMA (wave32): V_WMMA_F32_16X16X32_BF16
// ---------------------------------------------------------------------------
typedef __bf16 bf16_t;
typedef bf16_t v16bf __attribute__((ext_vector_type(16)));
typedef bf16_t v8bf  __attribute__((ext_vector_type(8)));
typedef float  v8f   __attribute__((ext_vector_type(8)));

#define LN_EPS 1e-6f

// Problem dims (fixed by the reference)
static constexpr int B_  = 64;
static constexpr int T_  = 256;
static constexpr int D_  = 1024;
static constexpr int K2  = 2048;          // input-part K (inputo ++ attn)
static constexpr int N4  = 4096;          // 4*d gate width
static constexpr int M1  = B_ * T_;       // 16384 rows in the big GEMM

__device__ __forceinline__ v16bf cat8(v8bf lo, v8bf hi) {
  return __builtin_shufflevector(lo, hi, 0,1,2,3,4,5,6,7,8,9,10,11,12,13,14,15);
}

// CDNA5 async global->LDS copy (16B per lane), tracked by ASYNCcnt.
// VDST = LDS byte offset (low 32 bits of the flat shared address), VADDR = u64.
__device__ __forceinline__ void async_b128(void* lds_ptr, const bf16_t* g) {
  uint32_t loff = (uint32_t)(uintptr_t)lds_ptr;
  unsigned long long ga = (unsigned long long)(uintptr_t)g;
  asm volatile("global_load_async_to_lds_b128 %0, %1, off"
               :: "v"(loff), "v"(ga)
               : "memory");
}

// ---------------------------------------------------------------------------
// Phase 0a: pack concat(inputo, attn) -> bf16 in WMMA A-fragment tile order.
// A tile (mt, kt) = 16(M) x 32(K); lane l holds M = l%16 and
//   elems 0..7  -> K = (l<16 ? 0 : 8)  + i
//   elems 8..15 -> K = (l<16 ? 16 : 24) + (i-8)
// Packed: off = (mt*64 + kt)*512 + l*16 + i  (per-lane 16 elems contiguous)
// ---------------------------------------------------------------------------
__global__ __launch_bounds__(256) void pack_x_kernel(
    const float* __restrict__ inputo, const float* __restrict__ attn,
    bf16_t* __restrict__ Xpk) {
  size_t idx = (size_t)blockIdx.x * 256 + threadIdx.x;     // [0, M1*K2)
  size_t m = idx >> 11;                                    // row (b*T + t)
  int    k = (int)(idx & 2047);
  int mt = (int)(m >> 4), mo = (int)(m & 15);
  int kt = k >> 5, ko = k & 31;
  int l, i;
  if (ko < 8)       { l = mo;      i = ko;      }
  else if (ko < 16) { l = 16 + mo; i = ko - 8;  }
  else if (ko < 24) { l = mo;      i = ko - 8;  }   // 8 + (ko-16)
  else              { l = 16 + mo; i = ko - 16; }   // 8 + (ko-24)
  float v = (k < D_) ? inputo[m * D_ + k] : attn[m * D_ + (k - D_)];
  Xpk[((size_t)(mt * 64 + kt)) * 512 + l * 16 + i] = (bf16_t)v;
}

// ---------------------------------------------------------------------------
// Phase 0b: pack W [3072, 4096] fp32 -> bf16 WMMA B-tile layout.
// B tile (kt, nt) = 32(K) x 16(N); lane l holds N = l%16, K = (i<8 ? 0 : 8) + (l<16 ? 0 : 8) + i.
// off = (kt*256 + nt)*512 + l*16 + i
// ---------------------------------------------------------------------------
__global__ __launch_bounds__(256) void pack_w_kernel(
    const float* __restrict__ W, bf16_t* __restrict__ Wpk) {
  size_t idx = (size_t)blockIdx.x * 256 + threadIdx.x;     // [0, 3072*4096)
  int K = (int)(idx >> 12);
  int N = (int)(idx & 4095);
  int kt = K >> 5, ko = K & 31;
  int nt = N >> 4, no = N & 15;
  int l  = ((ko >> 3) & 1) ? (16 + no) : no;
  int i  = (ko & 7) + ((ko >> 4) << 3);
  size_t off = ((size_t)(kt * 256 + nt)) * 512 + l * 16 + i;
  Wpk[off] = (bf16_t)W[idx];
}

// ---------------------------------------------------------------------------
// Phase 0c: broadcast initial state. cx fp32 [64,1024], h bf16 [64,1024].
// ---------------------------------------------------------------------------
__global__ __launch_bounds__(256) void init_state_kernel(
    const float* __restrict__ init_hx, const float* __restrict__ init_cx,
    float* __restrict__ cx, bf16_t* __restrict__ Hbf) {
  int idx = blockIdx.x * 256 + threadIdx.x;                // [0, 64*1024)
  int j = idx & (D_ - 1);
  cx[idx]  = init_cx[j];
  Hbf[idx] = (bf16_t)init_hx[j];
}

// ---------------------------------------------------------------------------
// Phase 1: Zx = X @ W[:2048,:] + b   (16384 x 2048 x 4096, bf16 WMMA)
// Workgroup (256 thr / 8 waves) -> 128x128 output tile.
// Wave grid 4(M) x 2(N): each wave 32x64 = 2x4 accumulators.
// Per K-step: async-stage A(8KB)+B(8KB) into LDS (double buffered),
// s_wait_asynccnt keeps next stage in flight while WMMAs consume current.
// grid: (16384/128) * (4096/128) = 128*32 = 4096 blocks.
// ---------------------------------------------------------------------------
__device__ __forceinline__ void zx_compute_step(
    const bf16_t* sAc, const bf16_t* sBc, int wm, int wn, int lane,
    v8f acc[2][4]) {
  v16bf a[2];
#pragma unroll
  for (int i = 0; i < 2; ++i) {
    const bf16_t* p = sAc + (wm * 2 + i) * 512 + lane * 16;
    a[i] = cat8(*(const v8bf*)p, *(const v8bf*)(p + 8));
  }
#pragma unroll
  for (int j = 0; j < 4; ++j) {
    const bf16_t* p = sBc + (wn * 4 + j) * 512 + lane * 16;
    v16bf b = cat8(*(const v8bf*)p, *(const v8bf*)(p + 8));
#pragma unroll
    for (int i = 0; i < 2; ++i) {
      acc[i][j] = __builtin_amdgcn_wmma_f32_16x16x32_bf16(
          false, a[i], false, b, (short)0, acc[i][j], false, false);
    }
  }
}

__global__ __launch_bounds__(256) void gemm_zx_kernel(
    const bf16_t* __restrict__ Xpk, const bf16_t* __restrict__ Wpk,
    const float* __restrict__ bias, float* __restrict__ Zx) {
  __shared__ bf16_t sA[2][4096];   // 8 A-tiles (128x32) per buffer
  __shared__ bf16_t sB[2][4096];   // 8 B-tiles (32x128) per buffer

  const int tid   = threadIdx.x;
  const int lane  = tid & 31;
  const int w     = tid >> 5;
  const int wm    = w >> 1;            // 0..3
  const int wn    = w & 1;             // 0..1
  const int half  = lane >> 4;
  const int lan16 = lane & 15;
  const int bm  = blockIdx.x >> 5;     // 0..127
  const int bn  = blockIdx.x & 31;     // 0..31
  const int mt0 = bm * 8;              // 8 M-tiles per WG
  const int nt0 = bn * 8;              // 8 N-tiles per WG
  constexpr int NK = K2 / 32;          // 64

  v8f acc[2][4] = {};

  // Per-thread staging: 2 chunks of A + 2 chunks of B (16B each), 512 chunks
  // cover 8KB.  A chunk c: tile = c>>6, intra = c&63.  B is 8KB contiguous.
  auto stage = [&](int kt, int buf) {
#pragma unroll
    for (int r = 0; r < 2; ++r) {
      int c = tid + r * 256;
      const bf16_t* gA =
          Xpk + ((size_t)((mt0 + (c >> 6)) * 64 + kt)) * 512 + (c & 63) * 8;
      async_b128(&sA[buf][c * 8], gA);
      const bf16_t* gB = Wpk + ((size_t)(kt * 256 + nt0)) * 512 + c * 8;
      async_b128(&sB[buf][c * 8], gB);
    }
  };

  stage(0, 0);
  for (int kt = 0; kt < NK - 1; ++kt) {
    const int cur = kt & 1;
    stage(kt + 1, cur ^ 1);                       // prefetch next K-slice
    asm volatile("s_wait_asynccnt 0x4" ::: "memory");  // current slice done
    __syncthreads();
    zx_compute_step(&sA[cur][0], &sB[cur][0], wm, wn, lane, acc);
    __syncthreads();                              // done reading before restage
  }
  asm volatile("s_wait_asynccnt 0x0" ::: "memory");
  __syncthreads();
  zx_compute_step(&sA[(NK - 1) & 1][0], &sB[(NK - 1) & 1][0], wm, wn, lane, acc);

  // C/D layout: VGPR e holds row M = half*8 + e, column N = lane%16
#pragma unroll
  for (int i = 0; i < 2; ++i) {
    const int mbase = (mt0 + wm * 2 + i) * 16 + half * 8;
#pragma unroll
    for (int j = 0; j < 4; ++j) {
      int n = (nt0 + wn * 4 + j) * 16 + lan16;
      float bn = bias[n];
#pragma unroll
      for (int e = 0; e < 8; ++e) {
        Zx[(size_t)(mbase + e) * N4 + n] = acc[i][j][e] + bn;
      }
    }
  }
}

// ---------------------------------------------------------------------------
// Phase 2a (per step t): z = Hbf @ W[2048:,:] + Zx[:, t, :]
// 64 x 1024 x 4096; 4 M-tiles * 64 N-groups = 256 waves = 32 blocks.
// Software-pipelined: next K-slice fragments load while current WMMAs issue.
// ---------------------------------------------------------------------------
__global__ __launch_bounds__(256) void gemm_zh_kernel(
    const bf16_t* __restrict__ Hbf, const bf16_t* __restrict__ Wpk,
    const float* __restrict__ Zx, float* __restrict__ zrow, int t) {
  const int wid   = blockIdx.x * 8 + (threadIdx.x >> 5);
  const int lane  = threadIdx.x & 31;
  const int half  = lane >> 4;
  const int lan16 = lane & 15;
  const int mt = wid >> 6;              // 0..3  (batch tile)
  const int ng = wid & 63;              // 0..63

  v8f acc[4] = {v8f{}, v8f{}, v8f{}, v8f{}};
  const int arow = mt * 16 + lan16;     // batch row 0..63
  const bf16_t* aptr = Hbf + (size_t)arow * D_ + half * 8;

  auto loadA = [&](int kt) {
    return cat8(*(const v8bf*)(aptr + kt * 32),
                *(const v8bf*)(aptr + kt * 32 + 16));
  };
  auto loadB = [&](int kt, int j) {
    // recurrent rows of W start at K=2048 -> packed kt offset 64
    const bf16_t* p =
        Wpk + ((size_t)((64 + kt) * 256 + (ng * 4 + j))) * 512 + lane * 16;
    return cat8(*(const v8bf*)p, *(const v8bf*)(p + 8));
  };

  v16bf a_cur = loadA(0);
  v16bf b_cur[4];
#pragma unroll
  for (int j = 0; j < 4; ++j) b_cur[j] = loadB(0, j);

  constexpr int NK = D_ / 32;           // 32
  for (int kt = 0; kt < NK; ++kt) {
    v16bf a_nxt = a_cur;
    v16bf b_nxt[4] = {b_cur[0], b_cur[1], b_cur[2], b_cur[3]};
    if (kt + 1 < NK) {
      a_nxt = loadA(kt + 1);
#pragma unroll
      for (int j = 0; j < 4; ++j) b_nxt[j] = loadB(kt + 1, j);
    }
#pragma unroll
    for (int j = 0; j < 4; ++j) {
      acc[j] = __builtin_amdgcn_wmma_f32_16x16x32_bf16(
          false, a_cur, false, b_cur[j], (short)0, acc[j], false, false);
    }
    a_cur = a_nxt;
#pragma unroll
    for (int j = 0; j < 4; ++j) b_cur[j] = b_nxt[j];
  }

  const int mbase = mt * 16 + half * 8;
#pragma unroll
  for (int j = 0; j < 4; ++j) {
    int n = (ng * 4 + j) * 16 + lan16;
#pragma unroll
    for (int e = 0; e < 8; ++e) {
      int b = mbase + e;                // batch row
      float zx = Zx[((size_t)b * T_ + t) * N4 + n];   // bias already folded in
      zrow[(size_t)b * N4 + n] = acc[j][e] + zx;
    }
  }
}

// ---------------------------------------------------------------------------
// Phase 2b (per step t): LayerNorm over 4096 + gates + state + residual out.
// One block per batch row; 256 threads.
// ---------------------------------------------------------------------------
__global__ __launch_bounds__(256) void ln_gates_kernel(
    const float* __restrict__ zrow, const float* __restrict__ lnw,
    const float* __restrict__ lnb, const float* __restrict__ inputo,
    float* __restrict__ cx, bf16_t* __restrict__ Hbf,
    float* __restrict__ out, int t) {
  const int b = blockIdx.x;
  const int tid = threadIdx.x;
  const float* zr = zrow + (size_t)b * N4;

  __shared__ float ssum[256];
  __shared__ float ssq[256];
  float s = 0.f, q = 0.f;
#pragma unroll
  for (int i = 0; i < 16; ++i) {
    float v = zr[tid + i * 256];
    s += v; q += v * v;
  }
  ssum[tid] = s; ssq[tid] = q;
  __syncthreads();
  for (int off = 128; off > 0; off >>= 1) {
    if (tid < off) { ssum[tid] += ssum[tid + off]; ssq[tid] += ssq[tid + off]; }
    __syncthreads();
  }
  const float mu  = ssum[0] * (1.0f / 4096.0f);
  const float var = ssq[0] * (1.0f / 4096.0f) - mu * mu;
  const float rs  = rsqrtf(var + LN_EPS);

#pragma unroll
  for (int r = 0; r < 4; ++r) {
    int j = tid + r * 256;              // 0..1023
    float zn[4];
#pragma unroll
    for (int g = 0; g < 4; ++g) {
      float zz = zr[g * D_ + j];
      zn[g] = (zz - mu) * rs * lnw[g * D_ + j] + lnb[g * D_ + j];
    }
    float ig = 1.0f / (1.0f + expf(-zn[0]));
    float fg = 1.0f / (1.0f + expf(-zn[1]));
    float og = 1.0f / (1.0f + expf(-zn[2]));
    float hid = 0.5f * zn[3] * (1.0f + erff(zn[3] * 0.70710678118654752f));

    int sidx = b * D_ + j;
    float cnew = fg * cx[sidx] + ig * hid;
    cx[sidx] = cnew;
    float h = og * cnew;
    Hbf[sidx] = (bf16_t)h;
    size_t oidx = ((size_t)b * T_ + t) * D_ + j;
    out[oidx] = h + inputo[oidx];
  }
}

// ---------------------------------------------------------------------------
// Host-side launcher
// ---------------------------------------------------------------------------
extern "C" void kernel_launch(void* const* d_in, const int* in_sizes, int n_in,
                              void* d_out, int out_size, void* d_ws, size_t ws_size,
                              hipStream_t stream) {
  const float* inputo  = (const float*)d_in[0];   // [64,256,1024]
  const float* attn    = (const float*)d_in[1];   // [64,256,1024]
  const float* W       = (const float*)d_in[2];   // [3072,4096]
  const float* bias    = (const float*)d_in[3];   // [4096]
  const float* lnw     = (const float*)d_in[4];   // [4,1024]
  const float* lnb     = (const float*)d_in[5];   // [4,1024]
  const float* init_hx = (const float*)d_in[6];   // [1,1024]
  const float* init_cx = (const float*)d_in[7];   // [1,1024]
  float* out = (float*)d_out;                     // [64,256,1024]

  // Workspace layout (256B aligned sections)
  char* ws = (char*)d_ws;
  const size_t off_Xpk = 0;                                      // 64 MiB
  const size_t off_Wpk = off_Xpk + (size_t)M1 * K2 * 2;          // +24 MiB
  const size_t off_Zx  = off_Wpk + (size_t)3072 * 4096 * 2;      // +256 MiB
  const size_t off_z   = off_Zx  + (size_t)M1 * N4 * 4;          // +1 MiB
  const size_t off_cx  = off_z   + (size_t)B_ * N4 * 4;          // +256 KiB
  const size_t off_hb  = off_cx  + (size_t)B_ * D_ * 4;          // +128 KiB

  bf16_t* Xpk  = (bf16_t*)(ws + off_Xpk);
  bf16_t* Wpk  = (bf16_t*)(ws + off_Wpk);
  float*  Zx   = (float*) (ws + off_Zx);
  float*  zrow = (float*) (ws + off_z);
  float*  cx   = (float*) (ws + off_cx);
  bf16_t* Hbf  = (bf16_t*)(ws + off_hb);

  // Phase 0: data packing / state init
  pack_x_kernel<<<(int)(((size_t)M1 * K2) / 256), 256, 0, stream>>>(inputo, attn, Xpk);
  pack_w_kernel<<<(int)(((size_t)3072 * 4096) / 256), 256, 0, stream>>>(W, Wpk);
  init_state_kernel<<<(B_ * D_) / 256, 256, 0, stream>>>(init_hx, init_cx, cx, Hbf);

  // Phase 1: big parallel GEMM for the input-dependent part of z
  gemm_zx_kernel<<<4096, 256, 0, stream>>>(Xpk, Wpk, bias, Zx);

  // Phase 2: sequential recurrence (graph-captured per-step kernels)
  for (int t = 0; t < T_; ++t) {
    gemm_zh_kernel<<<32, 256, 0, stream>>>(Hbf, Wpk, Zx, zrow, t);
    ln_gates_kernel<<<B_, 256, 0, stream>>>(zrow, lnw, lnb, inputo, cx, Hbf, out, t);
  }
}
